// SCAttention_13967233647492
// MI455X (gfx1250) — matrix operand order repaired and checked by reading, not compiled, not run on hardware
//
#include <hip/hip_runtime.h>

typedef __attribute__((ext_vector_type(16))) _Float16 v16h;
typedef __attribute__((ext_vector_type(2)))  _Float16 v2h;
typedef __attribute__((ext_vector_type(8)))  float    v8f;
typedef _Float16 half_t;

#define GN_EPS 1e-5f
// Flip to 0 if the assembler rejects the async-to-LDS mnemonic/operands.
#define USE_ASYNC_LDS 1

#if USE_ASYNC_LDS
// GLOBAL_LOAD_ASYNC_TO_LDS_B128, GV mode: VDST = LDS byte address (VGPR),
// VADDR = 64-bit global address (VGPR pair). Tracked by ASYNCcnt.
__device__ __forceinline__ void async_b128(unsigned lds_off, const void* gaddr) {
  asm volatile("global_load_async_to_lds_b128 %0, %1, off"
               :: "v"(lds_off), "v"(gaddr) : "memory");
}
__device__ __forceinline__ void wait_async0() {
  asm volatile("s_wait_asynccnt 0x0" ::: "memory");
}
// Generic shared-space pointer -> LDS byte offset (aperture: low 32 bits).
__device__ __forceinline__ unsigned lds_off_of(const void* p) {
  return (unsigned)(unsigned long long)p;
}
#endif

// Packed f16 max (2 rows per 32-bit lane) for the swizzle reduction tree.
__device__ __forceinline__ unsigned pkmax(unsigned a, unsigned b) {
  v2h x = __builtin_bit_cast(v2h, a), y = __builtin_bit_cast(v2h, b);
  v2h r;
  r[0] = (x[0] > y[0]) ? x[0] : y[0];
  r[1] = (x[1] > y[1]) ? x[1] : y[1];
  return __builtin_bit_cast(unsigned, r);
}

// ---------------- WMMA f16 16x16x32 fragment index helpers (wave32) --------
// A matrix 16x32 (MxK), 16-bit: lanes 0-15 => M=lane, K in {0..7,16..23};
// lanes 16-31 => M=lane-16, K in {8..15,24..31}. VGPR p packs K lo/hi pairs.
__device__ __forceinline__ int a_k_of(int lane, int e) {
  int p = e >> 1;
  int koff = (lane & 16) ? 8 : 0;
  int base = (p < 4) ? (2 * p) : (16 + 2 * (p - 4));
  return koff + base + (e & 1);
}
// B matrix 32x16 (KxN), 16-bit: lanes 0-15 => N=lane, K in {0..7,16..23};
// lanes 16-31 => N=lane-16, K in {8..15,24..31}.
__device__ __forceinline__ int b_k_of(int lane, int e) {
  int p = e >> 1;
  int koff = (lane & 16) ? 8 : 0;
  int base = (p < 4) ? (2 * p) : (16 + 2 * (p - 4));
  return koff + base + (e & 1);
}
// C/D 16x16 f32: VGPR r holds row r (lanes 0-15, N=lane) and row r+8
// (lanes 16-31, N=lane-16).

// ---------------- 3x3x3 SAME conv, 64->64 channels, 16^3 spatial -----------
__global__ __launch_bounds__(256)
void conv3d_k(const float* __restrict__ in, const float* __restrict__ w,
              const float* __restrict__ bias, float* __restrict__ out) {
  const int z = blockIdx.x;    // 0..15
  const int b = blockIdx.y;    // 0..3
  const int tid = threadIdx.x;
  const int y = tid >> 4, x = tid & 15;
  __shared__ float tile[16][3][18][18];   // 62,208 B
  float acc[64];
#pragma unroll
  for (int co = 0; co < 64; ++co) acc[co] = bias[co];
  for (int chunk = 0; chunk < 4; ++chunk) {
    const int cin0 = chunk * 16;
    for (int idx = tid; idx < 16 * 3 * 18 * 18; idx += 256) {
      int c  = idx / (3 * 18 * 18);
      int r  = idx % (3 * 18 * 18);
      int zz = r / (18 * 18);
      int rr = r % (18 * 18);
      int yy = rr / 18, xx = rr % 18;
      int gz = z + zz - 1, gy = yy - 1, gx = xx - 1;
      float v = 0.f;
      if (gz >= 0 && gz < 16 && gy >= 0 && gy < 16 && gx >= 0 && gx < 16)
        v = in[(((size_t)(b * 64 + cin0 + c) * 16 + gz) * 16 + gy) * 16 + gx];
      tile[c][zz][yy][xx] = v;
    }
    __syncthreads();
    for (int c = 0; c < 16; ++c) {
      for (int kk = 0; kk < 27; ++kk) {
        int dz = kk / 9, dy = (kk / 3) % 3, dx = kk % 3;
        float v = tile[c][dz][y + dy][x + dx];
        const float* wp = w + (cin0 + c) * 27 + kk;  // +co*64*27 per cout
#pragma unroll
        for (int co = 0; co < 64; ++co)
          acc[co] = fmaf(v, wp[co * 64 * 27], acc[co]);
      }
    }
    __syncthreads();
  }
#pragma unroll
  for (int co = 0; co < 64; ++co)
    out[(((size_t)(b * 64 + co) * 16 + z) * 16 + y) * 16 + x] = acc[co];
}

// ---------------- GroupNorm stats: one block per (b, group) ----------------
__global__ __launch_bounds__(256)
void gn_stats_k(const float* __restrict__ t, float* __restrict__ stats) {
  const int bg = blockIdx.x;           // 0..15 : b*4+g
  const int b = bg >> 2, g = bg & 3;
  const float* p = t + (size_t)(b * 64 + g * 16) * 4096;
  float s = 0.f, s2 = 0.f;
  for (int i = threadIdx.x; i < 16 * 4096; i += 256) {
    float v = p[i];
    s += v; s2 += v * v;
  }
  __shared__ float rs[256], rs2[256];
  rs[threadIdx.x] = s; rs2[threadIdx.x] = s2;
  __syncthreads();
  for (int off = 128; off > 0; off >>= 1) {
    if (threadIdx.x < off) {
      rs[threadIdx.x]  += rs[threadIdx.x + off];
      rs2[threadIdx.x] += rs2[threadIdx.x + off];
    }
    __syncthreads();
  }
  if (threadIdx.x == 0) {
    float mean = rs[0] * (1.f / 65536.f);
    float var  = rs2[0] * (1.f / 65536.f) - mean * mean;
    stats[bg * 2]     = mean;
    stats[bg * 2 + 1] = rsqrtf(var + GN_EPS);
  }
}

// ---------------- GroupNorm apply + ReLU (f32 out, optional f16 mirror) ----
__global__ __launch_bounds__(256)
void gn_apply_relu_k(const float* __restrict__ t, const float* __restrict__ stats,
                     const float* __restrict__ gamma, const float* __restrict__ beta,
                     float* __restrict__ o32, half_t* __restrict__ o16) {
  int i = blockIdx.x * 256 + threadIdx.x;      // < 1,048,576
  int c = (i >> 12) & 63;
  int b = i >> 18;
  int bg = b * 4 + (c >> 4);
  float mean = stats[bg * 2], inv = stats[bg * 2 + 1];
  float v = (t[i] - mean) * inv * gamma[c] + beta[c];
  v = fmaxf(v, 0.f);
  o32[i] = v;
  if (o16) o16[i] = (half_t)v;
}

// ---------------- q/k/v channel projections; outputs f16, attn layouts -----
// qt,kt: [b][n][16] ; vt: [b][n][64]
__global__ __launch_bounds__(256)
void qkv_proj_k(const float* __restrict__ xf,
                const float* __restrict__ wq, const float* __restrict__ bq,
                const float* __restrict__ wk, const float* __restrict__ bk,
                const float* __restrict__ wv, const float* __restrict__ bv,
                half_t* __restrict__ qt, half_t* __restrict__ kt,
                half_t* __restrict__ vt) {
  __shared__ float swq[1024], swk[1024], swv[4096], sb[96];
  for (int i = threadIdx.x; i < 1024; i += 256) { swq[i] = wq[i]; swk[i] = wk[i]; }
  for (int i = threadIdx.x; i < 4096; i += 256) swv[i] = wv[i];
  if (threadIdx.x < 16)       sb[threadIdx.x] = bq[threadIdx.x];
  else if (threadIdx.x < 32)  sb[threadIdx.x] = bk[threadIdx.x - 16];
  else if (threadIdx.x < 96)  sb[threadIdx.x] = bv[threadIdx.x - 32];
  __syncthreads();
  int gn = blockIdx.x * 256 + threadIdx.x;   // 0..16383 : b*4096+n
  int b = gn >> 12, n = gn & 4095;
  float xv[64];
#pragma unroll
  for (int c = 0; c < 64; ++c) xv[c] = xf[(size_t)(b * 64 + c) * 4096 + n];
#pragma unroll 4
  for (int o = 0; o < 16; ++o) {
    float aq = sb[o], ak = sb[16 + o];
    for (int c = 0; c < 64; ++c) {
      aq = fmaf(swq[o * 64 + c], xv[c], aq);
      ak = fmaf(swk[o * 64 + c], xv[c], ak);
    }
    qt[(size_t)gn * 16 + o] = (half_t)aq;
    kt[(size_t)gn * 16 + o] = (half_t)ak;
  }
#pragma unroll 4
  for (int o = 0; o < 64; ++o) {
    float av = sb[32 + o];
    for (int c = 0; c < 64; ++c) av = fmaf(swv[o * 64 + c], xv[c], av);
    vt[(size_t)gn * 64 + o] = (half_t)av;
  }
}

// ---------------- fused flash spatial attention ----------------------------
// Never materializes the 4x4096x4096 A matrix. 8 waves/WG, each wave owns 16
// query rows; K/V chunks (32 keys) double-buffered in LDS via async DMA so
// the copy of chunk i+1 overlaps compute on chunk i. Softmax row-sums ride
// the matrix pipe (all-ones B tile, 5th WMMA accumulator); row-max uses a
// packed-f16 ds_swizzle XOR tree (f16 precision is exact-enough for the
// stabilizing shift, which cancels in the normalization).
__global__ __launch_bounds__(256)
void flash_sattn_k(const half_t* __restrict__ qt, const half_t* __restrict__ kt,
                   const half_t* __restrict__ vt, float* __restrict__ sa) {
  const int b = blockIdx.y;
  const int wv_ = threadIdx.x >> 5;        // wave 0..7
  const int lane = threadIdx.x & 31;
  const int mbase = blockIdx.x * 128 + wv_ * 16;
  __shared__ __align__(16) half_t Ks[2][32][16];   // [buf][key][dim]  2 KB
  __shared__ __align__(16) half_t Vs[2][32][64];   // [buf][key][ch]   8 KB
  __shared__ float Ps[8][16][32];                  // per-wave exp(S) staging

  // Q A-fragment: rows = queries, K-dim = 16 head dims (zero-padded to 32)
  v16h qa;
  {
    const half_t* qp = qt + (size_t)(b * 4096 + mbase + (lane & 15)) * 16;
#pragma unroll
    for (int e = 0; e < 16; ++e) {
      int k = a_k_of(lane, e);
      qa[e] = (k < 16) ? qp[k] : (half_t)0.f;
    }
  }
  v16h ones;
#pragma unroll
  for (int e = 0; e < 16; ++e) ones[e] = (half_t)1.f;

  v8f O[5] = {v8f{}, v8f{}, v8f{}, v8f{}, v8f{}};  // O[4] = row-sum tile
  float mx[8];
#pragma unroll
  for (int r = 0; r < 8; ++r) mx[r] = -1e30f;

  const int colw = lane & 15;
  const int rowoff = (lane & 16) ? 8 : 0;

#if USE_ASYNC_LDS
  {   // pre-stage chunk 0 into buffer 0: 256 x 16B for V, threads 0-63 also K
    const char* vgp = (const char*)(vt + (size_t)(b * 4096) * 64);
    async_b128(lds_off_of(&Vs[0][0][0]) + threadIdx.x * 16,
               vgp + threadIdx.x * 16);
    if (threadIdx.x < 64) {
      const char* kgp = (const char*)(kt + (size_t)(b * 4096) * 16);
      async_b128(lds_off_of(&Ks[0][0][0]) + threadIdx.x * 16,
                 kgp + threadIdx.x * 16);
    }
  }
#endif

  for (int n0 = 0; n0 < 4096; n0 += 32) {
    const int cur = (n0 >> 5) & 1;
#if USE_ASYNC_LDS
    wait_async0();        // chunk n0's DMA complete (this wave's pieces)
    __syncthreads();      // all waves' pieces visible; prev reads of ^1 done
    if (n0 + 32 < 4096) { // kick off chunk n0+32 into the other buffer
      const int nb = cur ^ 1;
      const char* vgp = (const char*)(vt + (size_t)(b * 4096 + n0 + 32) * 64);
      async_b128(lds_off_of(&Vs[nb][0][0]) + threadIdx.x * 16,
                 vgp + threadIdx.x * 16);
      if (threadIdx.x < 64) {
        const char* kgp = (const char*)(kt + (size_t)(b * 4096 + n0 + 32) * 16);
        async_b128(lds_off_of(&Ks[nb][0][0]) + threadIdx.x * 16,
                   kgp + threadIdx.x * 16);
      }
    }
#else
    for (int i = threadIdx.x; i < 32 * 16; i += 256) {
      int key = i >> 4, d = i & 15;
      Ks[cur][key][d] = kt[(size_t)(b * 4096 + n0 + key) * 16 + d];
    }
    for (int i = threadIdx.x; i < 32 * 64; i += 256) {
      int key = i >> 6, c = i & 63;
      Vs[cur][key][c] = vt[(size_t)(b * 4096 + n0 + key) * 64 + c];
    }
    __syncthreads();
#endif

    // S tiles: [16 queries x 16 keys] twice (keys n0..+15, n0+16..+31)
    v16h kb0, kb1;
#pragma unroll
    for (int e = 0; e < 16; ++e) {
      int k = b_k_of(lane, e);             // head-dim index (padded)
      kb0[e] = (k < 16) ? Ks[cur][colw][k]      : (half_t)0.f;
      kb1[e] = (k < 16) ? Ks[cur][16 + colw][k] : (half_t)0.f;
    }
    v8f zc = {};
    v8f s0 = __builtin_amdgcn_wmma_f32_16x16x32_f16(false, qa, false, kb0,
                                                    (short)0, zc, false, false);
    v8f s1 = __builtin_amdgcn_wmma_f32_16x16x32_f16(false, qa, false, kb1,
                                                    (short)0, zc, false, false);

    // chunk row-max: pack row pairs to f16x2, XOR-swizzle tree over 16 lanes
    unsigned pmax[4];
#pragma unroll
    for (int j = 0; j < 4; ++j) {
      float m0 = fmaxf(s0[2 * j],     s1[2 * j]);
      float m1 = fmaxf(s0[2 * j + 1], s1[2 * j + 1]);
      pmax[j] = __builtin_bit_cast(unsigned, __builtin_amdgcn_cvt_pkrtz(m0, m1));
    }
#pragma unroll
    for (int j = 0; j < 4; ++j) {
      unsigned v = pmax[j];
      v = pkmax(v, (unsigned)__builtin_amdgcn_ds_swizzle((int)v, 0x041F));
      v = pkmax(v, (unsigned)__builtin_amdgcn_ds_swizzle((int)v, 0x081F));
      v = pkmax(v, (unsigned)__builtin_amdgcn_ds_swizzle((int)v, 0x101F));
      v = pkmax(v, (unsigned)__builtin_amdgcn_ds_swizzle((int)v, 0x201F));
      pmax[j] = v;
    }
    // online rescale + exp; write P to per-wave LDS staging
#pragma unroll
    for (int r = 0; r < 8; ++r) {
      v2h hh = __builtin_bit_cast(v2h, pmax[r >> 1]);
      float cm = (float)hh[r & 1];
      float mn = fmaxf(mx[r], cm);
      float sc = __expf(mx[r] - mn);
      mx[r] = mn;
#pragma unroll
      for (int t = 0; t < 5; ++t) O[t][r] *= sc;
      Ps[wv_][r + rowoff][colw]      = __expf(s0[r] - mn);
      Ps[wv_][r + rowoff][16 + colw] = __expf(s1[r] - mn);
    }
    __syncthreads();
    // P A-fragment (re-layout via LDS); accumulate P·V and P·1 (row sums)
    v16h pa;
#pragma unroll
    for (int e = 0; e < 16; ++e)
      pa[e] = (half_t)Ps[wv_][lane & 15][a_k_of(lane, e)];
#pragma unroll
    for (int t = 0; t < 4; ++t) {
      v16h vb;
#pragma unroll
      for (int e = 0; e < 16; ++e)
        vb[e] = Vs[cur][b_k_of(lane, e)][t * 16 + colw];
      O[t] = __builtin_amdgcn_wmma_f32_16x16x32_f16(false, pa, false, vb,
                                                    (short)0, O[t], false, false);
    }
    O[4] = __builtin_amdgcn_wmma_f32_16x16x32_f16(false, pa, false, ones,
                                                  (short)0, O[4], false, false);
  }
  // normalize by the WMMA-accumulated row sums and store sa[b][c][m]
#pragma unroll
  for (int r = 0; r < 8; ++r) {
    float inv = 1.f / O[4][r];
    int m = mbase + r + rowoff;
#pragma unroll
    for (int t = 0; t < 4; ++t)
      sa[(size_t)(b * 64 + t * 16 + colw) * 4096 + m] = O[t][r] * inv;
  }
}

// ---------------- channel attention: softmax(X·Xᵀ)·X -----------------------
__global__ __launch_bounds__(256)
void chan_attn_k(const half_t* __restrict__ xh, float* __restrict__ ca) {
  const int b = blockIdx.x;
  const int wv_ = threadIdx.x >> 5, lane = threadIdx.x & 31;
  __shared__ float  Gs[64][64];
  __shared__ half_t Ah[64][64];
  const half_t* xb = xh + (size_t)b * 64 * 4096;
  const int rowoff = (lane & 16) ? 8 : 0;
  // Gram matrix G[c][d], K = 4096
  for (int t = wv_; t < 16; t += 8) {
    int ci = t >> 2, di = t & 3;
    const half_t* ap = xb + (size_t)(ci * 16 + (lane & 15)) * 4096;
    const half_t* bp = xb + (size_t)(di * 16 + (lane & 15)) * 4096;
    v8f acc = {};
    for (int n0 = 0; n0 < 4096; n0 += 32) {
      v16h af, bf;
#pragma unroll
      for (int e = 0; e < 16; ++e) {
        af[e] = ap[n0 + a_k_of(lane, e)];
        bf[e] = bp[n0 + b_k_of(lane, e)];
      }
      acc = __builtin_amdgcn_wmma_f32_16x16x32_f16(false, af, false, bf,
                                                   (short)0, acc, false, false);
    }
#pragma unroll
    for (int r = 0; r < 8; ++r)
      Gs[ci * 16 + r + rowoff][di * 16 + (lane & 15)] = acc[r];
  }
  __syncthreads();
  // row softmax (64 rows, one thread each)
  if (threadIdx.x < 64) {
    int row = threadIdx.x;
    float mxv = -1e30f;
    for (int d = 0; d < 64; ++d) mxv = fmaxf(mxv, Gs[row][d]);
    float s = 0.f;
    for (int d = 0; d < 64; ++d) s += __expf(Gs[row][d] - mxv);
    float inv = 1.f / s;
    for (int d = 0; d < 64; ++d)
      Ah[row][d] = (half_t)(__expf(Gs[row][d] - mxv) * inv);
  }
  __syncthreads();
  // ca = Ah(64x64) @ X(64x4096): 4 c-tiles x 256 n-tiles, K = 64 (2 steps)
  for (int j = wv_; j < 1024; j += 8) {
    int ci = j & 3, n0 = (j >> 2) * 16;
    v8f acc = {};
#pragma unroll
    for (int k0 = 0; k0 < 64; k0 += 32) {
      v16h af, bf;
#pragma unroll
      for (int e = 0; e < 16; ++e) {
        af[e] = Ah[ci * 16 + (lane & 15)][k0 + a_k_of(lane, e)];
        bf[e] = xb[(size_t)(k0 + b_k_of(lane, e)) * 4096 + n0 + (lane & 15)];
      }
      acc = __builtin_amdgcn_wmma_f32_16x16x32_f16(false, af, false, bf,
                                                   (short)0, acc, false, false);
    }
#pragma unroll
    for (int r = 0; r < 8; ++r)
      ca[(size_t)(b * 64 + ci * 16 + r + rowoff) * 4096 + n0 + (lane & 15)] = acc[r];
  }
}

// ---------------- y = x + sa + ca ------------------------------------------
__global__ __launch_bounds__(256)
void add3_k(const float* __restrict__ a, const float* __restrict__ b,
            const float* __restrict__ c, float* __restrict__ y) {
  int i = blockIdx.x * 256 + threadIdx.x;
  y[i] = a[i] + b[i] + c[i];
}

// ---------------- host orchestration ---------------------------------------
extern "C" void kernel_launch(void* const* d_in, const int* in_sizes, int n_in,
                              void* d_out, int out_size, void* d_ws, size_t ws_size,
                              hipStream_t stream) {
  (void)in_sizes; (void)n_in; (void)out_size; (void)ws_size;
  const float* x       = (const float*)d_in[0];
  const float* w_i     = (const float*)d_in[1];
  const float* b_i     = (const float*)d_in[2];
  const float* gamma_i = (const float*)d_in[3];
  const float* beta_i  = (const float*)d_in[4];
  const float* w_q     = (const float*)d_in[5];
  const float* b_q     = (const float*)d_in[6];
  const float* w_k     = (const float*)d_in[7];
  const float* b_k     = (const float*)d_in[8];
  const float* w_v     = (const float*)d_in[9];
  const float* b_v     = (const float*)d_in[10];
  const float* w_o     = (const float*)d_in[11];
  const float* b_o     = (const float*)d_in[12];
  const float* gamma_o = (const float*)d_in[13];
  const float* beta_o  = (const float*)d_in[14];

  char* ws = (char*)d_ws;
  size_t off = 0;
  auto alloc = [&](size_t bytes) {
    void* p = ws + off;
    off = (off + bytes + 255) & ~(size_t)255;
    return p;
  };
  const size_t NEL = 4ull * 64 * 4096;          // 1,048,576 voxels
  float*  t0  = (float*)alloc(NEL * 4);         // conv outputs (reused)
  float*  x1  = (float*)alloc(NEL * 4);         // post GN+ReLU (f32)
  half_t* x1h = (half_t*)alloc(NEL * 2);        // post GN+ReLU (f16)
  half_t* qt  = (half_t*)alloc(4ull * 4096 * 16 * 2);
  half_t* kt  = (half_t*)alloc(4ull * 4096 * 16 * 2);
  half_t* vt  = (half_t*)alloc(4ull * 4096 * 64 * 2);
  float*  sa  = (float*)alloc(NEL * 4);
  float*  ca  = (float*)alloc(NEL * 4);
  float*  yb  = (float*)alloc(NEL * 4);
  float*  st  = (float*)alloc(64 * 4);          // GN mean/invstd

  dim3 cg(16, 4);
  conv3d_k<<<cg, 256, 0, stream>>>(x, w_i, b_i, t0);
  gn_stats_k<<<16, 256, 0, stream>>>(t0, st);
  gn_apply_relu_k<<<4096, 256, 0, stream>>>(t0, st, gamma_i, beta_i, x1, x1h);
  qkv_proj_k<<<64, 256, 0, stream>>>(x1, w_q, b_q, w_k, b_k, w_v, b_v, qt, kt, vt);
  flash_sattn_k<<<dim3(32, 4), 256, 0, stream>>>(qt, kt, vt, sa);
  chan_attn_k<<<4, 256, 0, stream>>>(x1h, ca);
  add3_k<<<4096, 256, 0, stream>>>(x1, sa, ca, yb);
  conv3d_k<<<cg, 256, 0, stream>>>(yb, w_o, b_o, t0);
  gn_stats_k<<<16, 256, 0, stream>>>(t0, st);
  gn_apply_relu_k<<<4096, 256, 0, stream>>>(t0, st, gamma_o, beta_o,
                                            (float*)d_out, (half_t*)nullptr);
}
